// ChannelLinearAttention_17944373362866
// MI455X (gfx1250) — compile-verified, matrix operand order checked by reading, not run on hardware
//
#include <hip/hip_runtime.h>
#include <hip/hip_bf16.h>
#include <stdint.h>

// Problem constants (fixed by the reference's setup_inputs)
#define BB 16
#define CC 512
#define NN 4096
#define EPSF 1e-6f

typedef __attribute__((ext_vector_type(16))) __bf16        v16bf;
typedef __attribute__((ext_vector_type(8)))  float         v8f;
typedef __attribute__((ext_vector_type(4)))  float         f32x4;
typedef __attribute__((ext_vector_type(4)))  unsigned int  u32x4;
typedef __attribute__((ext_vector_type(4)))  unsigned short u16x4;

__device__ __forceinline__ unsigned short f2bf(float x) {
    unsigned int u = __float_as_uint(x);
    // round-to-nearest-even bf16
    u = (u + 0x7fffu + ((u >> 16) & 1u)) >> 16;
    return (unsigned short)u;
}

union FragU { v16bf h; u32x4 q[2]; };

// Load a 16x32 bf16 WMMA fragment row-slice: per ISA, lane's 16 halves are two
// contiguous 8-half chunks at [k0 + 8*laneHi, +8) and [k0+16+8*laneHi, +8).
__device__ __forceinline__ v16bf load_frag_global(const unsigned short* rowPtrAtK0, int laneHi) {
    FragU f;
    f.q[0] = *(const u32x4*)(rowPtrAtK0 + laneHi * 8);
    f.q[1] = *(const u32x4*)(rowPtrAtK0 + 16 + laneHi * 8);
    return f.h;
}

// Store one 16x16 f32 accumulator tile as bf16 (D layout: lane -> N = l&15,
// VGPR r -> M = r + 8*laneHi), row stride CC.
__device__ __forceinline__ void store_tile_bf(unsigned short* base, v8f acc, int laneHi) {
    #pragma unroll
    for (int r = 0; r < 8; ++r)
        base[(size_t)(r + 8 * laneHi) * CC] = f2bf(acc[r]);
}

// ---------------------------------------------------------------------------
// K1: per-(b,n) column stats over C: inv_norm, value_sum, w = inv*(inv*sum+eps)
// ---------------------------------------------------------------------------
__global__ void cla_colstats(const float* __restrict__ x,
                             float* __restrict__ inv_norm,
                             float* __restrict__ value_sum,
                             float* __restrict__ wbuf) {
    int gid = blockIdx.x * blockDim.x + threadIdx.x;   // over B*N
    int b = gid >> 12;            // / NN
    int n = gid & (NN - 1);
    const float* col = x + (size_t)b * CC * NN + n;
    float s = 0.f, ss = 0.f;
    #pragma unroll 4
    for (int c = 0; c < CC; ++c) {
        float v = col[(size_t)c * NN];
        s += v;
        ss = fmaf(v, v, ss);
    }
    float inv = rsqrtf(ss);                // 1 / ||col||
    float qcs = fmaf(inv, s, EPSF);        // q_colsum
    inv_norm[gid]  = inv;
    value_sum[gid] = s;
    wbuf[gid]      = inv * qcs;            // weight for tailor GEMV
}

// ---------------------------------------------------------------------------
// K2: convert x -> bf16 V and bf16 Qn = x * inv_norm[n] (float4 vectorized)
// ---------------------------------------------------------------------------
__global__ void cla_convert(const float* __restrict__ x,
                            const float* __restrict__ inv_norm,
                            unsigned short* __restrict__ Vbf,
                            unsigned short* __restrict__ Qbf) {
    int gid = blockIdx.x * blockDim.x + threadIdx.x;   // over B*C*N/4
    int base = gid << 2;
    int n  = base & (NN - 1);
    int bc = base >> 12;
    int b  = bc >> 9;              // / CC
    f32x4 xv = ((const f32x4*)x)[gid];
    f32x4 iv = *(const f32x4*)(inv_norm + (size_t)b * NN + n);
    u16x4 vb, qb;
    #pragma unroll
    for (int j = 0; j < 4; ++j) {
        vb[j] = f2bf(xv[j]);
        qb[j] = f2bf(xv[j] * iv[j]);
    }
    ((u16x4*)Vbf)[gid] = vb;
    ((u16x4*)Qbf)[gid] = qb;
}

// ---------------------------------------------------------------------------
// K3: tailor GEMV: t[b,c] = 1 / (N + sum_n x[b,c,n] * w[b,n])
// ---------------------------------------------------------------------------
__global__ void cla_tailor(const float* __restrict__ x,
                           const float* __restrict__ wbuf,
                           float* __restrict__ tbuf) {
    __shared__ float red[256];
    int b = blockIdx.x >> 9;       // / CC
    int c = blockIdx.x & (CC - 1);
    const float* xr = x + ((size_t)(b * CC + c)) * NN;
    const float* wr = wbuf + (size_t)b * NN;
    float p = 0.f;
    for (int n = threadIdx.x; n < NN; n += 256)
        p = fmaf(xr[n], wr[n], p);
    red[threadIdx.x] = p;
    __syncthreads();
    for (int s = 128; s > 0; s >>= 1) {
        if (threadIdx.x < s) red[threadIdx.x] += red[threadIdx.x + s];
        __syncthreads();
    }
    if (threadIdx.x == 0)
        tbuf[blockIdx.x] = 1.0f / ((float)NN + red[0]);
}

// ---------------------------------------------------------------------------
// K4: GEMM1 — matrixbf[b] = bf16( Vbf[b] @ Qbfᵀ[b] ), C x C, K = N = 4096.
// 2x2 register blocking: each wave computes a 32x32 output block (4 WMMA
// tiles). Each fragment load feeds two v_wmma_f32_16x16x32_bf16, doubling
// arithmetic intensity vs 1 tile/wave; operands stream from L2 (per-batch
// bf16 working set is 8 MiB << 192 MB L2).
// ---------------------------------------------------------------------------
__global__ void cla_gemm1(const unsigned short* __restrict__ Vbf,
                          const unsigned short* __restrict__ Qbf,
                          unsigned short* __restrict__ matrixbf) {
    int wave   = threadIdx.x >> 5;
    int l      = threadIdx.x & 31;
    int laneHi = l >> 4;
    int ln     = l & 15;
    int wg = blockIdx.x * 8 + wave;    // over B * 16 * 16 super-tiles (32x32)
    int b  = wg >> 8;
    int rm = wg & 255;
    int c0 = (rm >> 4) << 5;
    int m0 = (rm & 15) << 5;

    const unsigned short* aRow0 = Vbf + ((size_t)(b * CC + c0 + ln)) * NN;
    const unsigned short* aRow1 = aRow0 + (size_t)16 * NN;
    const unsigned short* bRow0 = Qbf + ((size_t)(b * CC + m0 + ln)) * NN;
    const unsigned short* bRow1 = bRow0 + (size_t)16 * NN;

    v8f acc00 = {0.f,0.f,0.f,0.f,0.f,0.f,0.f,0.f};
    v8f acc01 = acc00, acc10 = acc00, acc11 = acc00;
    #pragma unroll 2
    for (int k0 = 0; k0 < NN; k0 += 32) {
        __builtin_prefetch(aRow0 + k0 + 256, 0, 1);   // global_prefetch_b8
        __builtin_prefetch(bRow0 + k0 + 256, 0, 1);
        v16bf a0 = load_frag_global(aRow0 + k0, laneHi);
        v16bf a1 = load_frag_global(aRow1 + k0, laneHi);
        v16bf b0 = load_frag_global(bRow0 + k0, laneHi);
        v16bf b1 = load_frag_global(bRow1 + k0, laneHi);
        acc00 = __builtin_amdgcn_wmma_f32_16x16x32_bf16(false, a0, false, b0, (short)0, acc00, false, false);
        acc01 = __builtin_amdgcn_wmma_f32_16x16x32_bf16(false, a0, false, b1, (short)0, acc01, false, false);
        acc10 = __builtin_amdgcn_wmma_f32_16x16x32_bf16(false, a1, false, b0, (short)0, acc10, false, false);
        acc11 = __builtin_amdgcn_wmma_f32_16x16x32_bf16(false, a1, false, b1, (short)0, acc11, false, false);
    }
    unsigned short* o = matrixbf + ((size_t)(b * CC + c0)) * CC + m0 + ln;
    store_tile_bf(o, acc00, laneHi);
    store_tile_bf(o + 16, acc01, laneHi);
    store_tile_bf(o + (size_t)16 * CC, acc10, laneHi);
    store_tile_bf(o + (size_t)16 * CC + 16, acc11, laneHi);
}

// ---------------------------------------------------------------------------
// K5: GEMM2 + fused epilogue.
//   D[c,n] = sum_m matrixbf[c,m] * Qbf[m,n]           (K = C = 512)
//   out    = x + gamma * t[b,c] * (value_sum[b,n] + D)
// Workgroup = 8 waves, each computing 32 c-rows x 16 n-cols (2x1 blocking):
// the LDS B fragment feeds two WMMAs, and the 16 KB staged tile
// (Qbf[:,n0:n0+16], pre-shuffled into WMMA fragment order -> two ds_load_b128
// per fragment) is reused across 256 output rows.
// ---------------------------------------------------------------------------
__global__ void cla_gemm2_epi(const unsigned short* __restrict__ matrixbf,
                              const unsigned short* __restrict__ Qbf,
                              const float* __restrict__ x,
                              const float* __restrict__ value_sum,
                              const float* __restrict__ tbuf,
                              const float* __restrict__ gammap,
                              float* __restrict__ out) {
    __shared__ unsigned short ldsB[CC * 16];   // 16 KB, fragment-ordered

    int bid = blockIdx.x;                 // B * (N/16) * (C/256) blocks
    int cb  = bid & 1;
    int nt  = (bid >> 1) & 255;
    int b   = bid >> 9;
    int n0  = nt << 4;

    // Stage + shuffle: (k, n) -> lds[((s*32 + lane)<<4) + h], where
    // s = k>>5, kk = k&31, g = kk>>4, r = kk&15, laneHi = r>>3, h = g*8+(r&7),
    // lane = n + 16*laneHi  — inverse of the A/B fragment element mapping.
    for (int e = threadIdx.x; e < CC * 16; e += 256) {
        int k = e >> 4;
        int n = e & 15;
        unsigned short v = Qbf[((size_t)(b * CC + k)) * NN + n0 + n];
        int s  = k >> 5;
        int kk = k & 31;
        int g  = kk >> 4;
        int r  = kk & 15;
        int hi = r >> 3;
        int h  = g * 8 + (r & 7);
        ldsB[(((s << 5) + n + (hi << 4)) << 4) + h] = v;
    }
    __syncthreads();

    int wave   = threadIdx.x >> 5;
    int l      = threadIdx.x & 31;
    int laneHi = l >> 4;
    int ln     = l & 15;
    int c0     = cb * 256 + wave * 32;

    const unsigned short* aRow0 = matrixbf + ((size_t)(b * CC + c0 + ln)) * CC;
    const unsigned short* aRow1 = aRow0 + (size_t)16 * CC;

    v8f acc0 = {0.f,0.f,0.f,0.f,0.f,0.f,0.f,0.f};
    v8f acc1 = acc0;
    #pragma unroll
    for (int s = 0; s < CC / 32; ++s) {
        v16bf a0 = load_frag_global(aRow0 + s * 32, laneHi);
        v16bf a1 = load_frag_global(aRow1 + s * 32, laneHi);
        FragU bf;
        const u32x4* bp = (const u32x4*)&ldsB[((s << 5) + l) << 4];
        bf.q[0] = bp[0];
        bf.q[1] = bp[1];
        acc0 = __builtin_amdgcn_wmma_f32_16x16x32_bf16(false, a0, false, bf.h, (short)0, acc0, false, false);
        acc1 = __builtin_amdgcn_wmma_f32_16x16x32_bf16(false, a1, false, bf.h, (short)0, acc1, false, false);
    }

    float gamma = gammap[0];
    float vs    = value_sum[(size_t)b * NN + n0 + ln];   // per-lane n constant
    #pragma unroll
    for (int half = 0; half < 2; ++half) {
        v8f acc = half ? acc1 : acc0;
        int cbase = c0 + half * 16;
        #pragma unroll
        for (int r = 0; r < 8; ++r) {
            int c = cbase + r + 8 * laneHi;
            size_t idx = ((size_t)(b * CC + c)) * NN + n0 + ln;
            float t = tbuf[b * CC + c];
            out[idx] = fmaf(gamma * t, vs + acc[r], x[idx]);
        }
    }
}

// ---------------------------------------------------------------------------
extern "C" void kernel_launch(void* const* d_in, const int* in_sizes, int n_in,
                              void* d_out, int out_size, void* d_ws, size_t ws_size,
                              hipStream_t stream) {
    const float* x     = (const float*)d_in[0];
    const float* gamma = (const float*)d_in[1];
    float* out = (float*)d_out;

    // Workspace layout:
    //   [0]          inv_norm   B*N f32          (256 KB)
    //   [+B*N]       value_sum  B*N f32          (256 KB)
    //   [+2*B*N]     wbuf       B*N f32          (256 KB)
    //   [+3*B*N]     tbuf       B*C f32          ( 32 KB)
    //   halves @ 1 MiB offset:
    //     matrixbf   B*C*C bf16 ( 8 MiB), Vbf B*C*N bf16 (64 MiB), Qbf (64 MiB)
    float* inv_norm  = (float*)d_ws;
    float* value_sum = inv_norm + BB * NN;
    float* wbuf      = value_sum + BB * NN;
    float* tbuf      = wbuf + BB * NN;
    unsigned short* matrixbf = (unsigned short*)((char*)d_ws + (1u << 20));
    unsigned short* Vbf      = matrixbf + (size_t)BB * CC * CC;
    unsigned short* Qbf      = Vbf + (size_t)BB * CC * NN;

    // K1: column stats
    cla_colstats<<<(BB * NN) / 256, 256, 0, stream>>>(x, inv_norm, value_sum, wbuf);
    // K2: bf16 conversion of V and Qn
    cla_convert<<<((size_t)BB * CC * NN / 4) / 256, 256, 0, stream>>>(x, inv_norm, Vbf, Qbf);
    // K3: tailor_sum GEMV
    cla_tailor<<<BB * CC, 256, 0, stream>>>(x, wbuf, tbuf);
    // K4: matrix = V @ Qnᵀ  (bf16 WMMA, 2x2 blocked, K = 4096)
    cla_gemm1<<<(BB * 16 * 16) / 8, 256, 0, stream>>>(Vbf, Qbf, matrixbf);
    // K5: matrix @ Qn + fused epilogue -> out (2x1 blocked)
    cla_gemm2_epi<<<BB * (NN / 16) * (CC / 256), 256, 0, stream>>>(
        matrixbf, Qbf, x, value_sum, tbuf, gamma, out);
}
